// GCN_STRC_16604343566789
// MI455X (gfx1250) — matrix operand order repaired and checked by reading, not compiled, not run on hardware
//
#include <hip/hip_runtime.h>
#include <hip/hip_bf16.h>

typedef __attribute__((ext_vector_type(16))) __bf16 v16bf;
typedef __attribute__((ext_vector_type(8)))  __bf16 v8bf;
typedef __attribute__((ext_vector_type(8)))  float  v8f;

#define EPSV 1e-5f
#define BSTRIDE 40  // padded LDS k-stride (bf16 elems): 80B, 16B-aligned, conflict-free

// ---------------- small utility kernels ----------------
__global__ void k_fill(float* __restrict__ p, float v, int n) {
  int i = blockIdx.x * blockDim.x + threadIdx.x;
  if (i < n) p[i] = v;
}

__global__ void k_count_deg(const int* __restrict__ col, float* __restrict__ deg, int E) {
  int i = blockIdx.x * blockDim.x + threadIdx.x;
  if (i < E) atomicAdd(&deg[col[i]], 1.0f);
}

__global__ void k_rsqrt_inplace(float* __restrict__ p, int n) {
  int i = blockIdx.x * blockDim.x + threadIdx.x;
  if (i < n) p[i] = rsqrtf(p[i]);
}

// f32 -> bf16 elementwise
__global__ void k_cvt_bf16(const float* __restrict__ in, __bf16* __restrict__ out, int n) {
  int i = blockIdx.x * blockDim.x + threadIdx.x;
  if (i < n) out[i] = (__bf16)in[i];
}

// W[K,F] f32 -> Wt[F,K] bf16 (transposed, so B fragments are contiguous in K)
__global__ void k_transpose_bf16(const float* __restrict__ W, __bf16* __restrict__ Wt,
                                 int K, int F) {
  int i = blockIdx.x * blockDim.x + threadIdx.x;  // i = f*K + k
  if (i >= K * F) return;
  int f = i / K, k = i - f * K;
  Wt[i] = (__bf16)W[(size_t)k * F + f];
}

// ---------------- WMMA GEMM: C[M,Fout] = A[M,K] @ B[K,Fout] --------------------
// A: bf16 row-major. Bt: bf16 [Fout,K] (transposed). f32 accumulate.
// block = 256 threads = 8 wave32. Wave tile = 16x64 (4 accumulators, A reused 4x).
// B tile (64 cols x 32 k) double-buffered in LDS via async global->LDS copies
// (ASYNCcnt), overlapping next-tile fetch with current-tile WMMAs; 1 barrier/step.
__global__ __launch_bounds__(256) void k_gemm_bf16_wmma(
    const __bf16* __restrict__ A, const __bf16* __restrict__ Bt,
    float* __restrict__ C, int M, int K, int Fout) {
  __shared__ __bf16 Bs[2][64 * BSTRIDE];
  const int lane = threadIdx.x & 31;
  const int h  = lane >> 4;   // half-wave select
  const int ln = lane & 15;
  const int m0 = (blockIdx.x * 8 + (threadIdx.x >> 5)) * 16;
  const int n0 = blockIdx.y * 64;
  // M is a multiple of 16: wave tiles are fully in range or fully out (clamp OOB).
  const int mrow = (m0 < M) ? (m0 + ln) : ln;
  const __bf16* arow = A + (size_t)mrow * K;

  // staging: thread t -> col sc = t>>2 (0..63), 8-elem segment sseg = (t&3)*8
  const int sc = threadIdx.x >> 2;
  const int sseg = (threadIdx.x & 3) * 8;
  const __bf16* bsrc = Bt + (size_t)(n0 + sc) * K + sseg;
  // LDS byte addresses (flat-aperture rule: LDS offset = addr[31:0])
  unsigned lds0 = (unsigned)(size_t)&Bs[0][sc * BSTRIDE + sseg];
  unsigned lds1 = (unsigned)(size_t)&Bs[1][sc * BSTRIDE + sseg];

  // prologue: async-stage first B tile into buffer 0
  asm volatile("global_load_async_to_lds_b128 %0, %1, off"
               :: "v"(lds0), "v"(bsrc) : "memory");

  v8f acc0 = {}, acc1 = {}, acc2 = {}, acc3 = {};
  int p = 0;
  for (int k0 = 0; k0 < K; k0 += 32) {
    // current tile resident; previous tile's LDS reads already consumed by WMMAs
    asm volatile("s_wait_asynccnt 0x0" ::: "memory");
    __syncthreads();
    if (k0 + 32 < K) {  // overlap: fetch next tile into the other buffer
      const __bf16* nsrc = bsrc + k0 + 32;
      unsigned nlds = p ? lds0 : lds1;
      asm volatile("global_load_async_to_lds_b128 %0, %1, off"
                   :: "v"(nlds), "v"(nsrc) : "memory");
      __builtin_prefetch(arow + k0 + 32, 0, 0);
    }

    // A fragment, CDNA5 16-bit 16x32 layout: lane 16h+m,
    // elems 0..7 -> K = k0+h*8+0..7 ; elems 8..15 -> K = k0+16+h*8+0..7
    v8bf alo = *(const v8bf*)(arow + k0 + h * 8);
    v8bf ahi = *(const v8bf*)(arow + k0 + 16 + h * 8);
    v16bf a = __builtin_shufflevector(alo, ahi, 0, 1, 2, 3, 4, 5, 6, 7,
                                      8, 9, 10, 11, 12, 13, 14, 15);

    // B fragment 32x16: lane 16h+n, elems 0..7 -> K = k0+h*8+0..7 ; elems 8..15 -> K = k0+16+h*8+0..7
    const __bf16* bbase = &Bs[p][0];
#define DO_NT(nt, accv)                                                          \
    {                                                                            \
      const __bf16* bp = bbase + ((nt) * 16 + ln) * BSTRIDE + h * 8;             \
      v8bf blo = *(const v8bf*)bp;                                               \
      v8bf bhi = *(const v8bf*)(bp + 16);                                        \
      v16bf b = __builtin_shufflevector(blo, bhi, 0, 1, 2, 3, 4, 5, 6, 7,        \
                                        8, 9, 10, 11, 12, 13, 14, 15);           \
      accv = __builtin_amdgcn_wmma_f32_16x16x32_bf16(false, a, false, b,         \
                                                     (short)0, accv, false, false); \
    }
    DO_NT(0, acc0)
    DO_NT(1, acc1)
    DO_NT(2, acc2)
    DO_NT(3, acc3)
#undef DO_NT
    p ^= 1;
  }

  if (m0 < M) {
    // C/D layout: elem r -> row h*8+r, col = ln (+16 per n-tile)
#pragma unroll
    for (int r = 0; r < 8; ++r) {
      float* cr = C + (size_t)(m0 + h * 8 + r) * Fout + n0 + ln;
      cr[0]  = acc0[r];
      cr[16] = acc1[r];
      cr[32] = acc2[r];
      cr[48] = acc3[r];
    }
  }
}

// ---------------- self-loop + bias init: out[i,f] = bias[f] + h[i,f]*dis[i]^2 -------
__global__ void k_self_bias(const float* __restrict__ h, const float* __restrict__ dis,
                            const float* __restrict__ bias, float* __restrict__ out,
                            int total, int F) {
  int i = blockIdx.x * blockDim.x + threadIdx.x;
  if (i >= total) return;
  int node = i / F, f = i - node * F;
  float d = dis[node];
  out[i] = bias[f] + h[i] * d * d;
}

// ---------------- edge scatter-add: out[dIdx[e]] += w * src[sIdx[e]] ---------------
// one wave per edge; float4 gathers, float atomics. dis==nullptr -> w=1 (raw adjacency)
__global__ void k_edge_scatter(const float* __restrict__ src, const int* __restrict__ sIdx,
                               const int* __restrict__ dIdx, const float* __restrict__ dis,
                               float* __restrict__ out, int E, int F) {
  int e = (blockIdx.x * blockDim.x + threadIdx.x) >> 5;
  int lane = threadIdx.x & 31;
  if (e >= E) return;
  int s = sIdx[e], d = dIdx[e];
  float w = dis ? dis[s] * dis[d] : 1.0f;
  const float4* sp = (const float4*)(src + (size_t)s * F);
  float* op = out + (size_t)d * F;
  int nv = F >> 2;
  for (int i = lane; i < nv; i += 32) {
    float4 v = sp[i];
    atomicAdd(op + 4 * i + 0, v.x * w);
    atomicAdd(op + 4 * i + 1, v.y * w);
    atomicAdd(op + 4 * i + 2, v.z * w);
    atomicAdd(op + 4 * i + 3, v.w * w);
  }
}

// ---------------- BatchNorm (batch stats over node dim) ---------------------------
__global__ void k_bn_partial(const float* __restrict__ x, float* __restrict__ s,
                             float* __restrict__ s2, int Nrows, int F, int rchunk) {
  int c = blockIdx.x * blockDim.x + threadIdx.x;
  if (c >= F) return;
  int r0 = blockIdx.y * rchunk;
  int r1 = min(r0 + rchunk, Nrows);
  float a = 0.f, b = 0.f;
  for (int r = r0; r < r1; ++r) {
    float v = x[(size_t)r * F + c];
    a += v; b += v * v;
  }
  atomicAdd(&s[c], a);
  atomicAdd(&s2[c], b);
}

__global__ void k_bn_finalize(const float* __restrict__ s, const float* __restrict__ s2,
                              float* __restrict__ mu, float* __restrict__ rs,
                              int Nrows, int F) {
  int c = blockIdx.x * blockDim.x + threadIdx.x;
  if (c >= F) return;
  float inv = 1.0f / (float)Nrows;
  float m = s[c] * inv;
  float var = s2[c] * inv - m * m;
  mu[c] = m;
  rs[c] = rsqrtf(fmaxf(var, 0.f) + EPSV);
}

// normalize; optional f32 dst, optional bf16 dst (feeds next WMMA GEMM), optional acc
__global__ void k_bn_apply(const float* __restrict__ x, const float* __restrict__ mu,
                           const float* __restrict__ rs, const float* __restrict__ gamma,
                           const float* __restrict__ beta, float* __restrict__ dst,
                           __bf16* __restrict__ bdst, float* __restrict__ acc,
                           int total, int F, int relu) {
  int i = blockIdx.x * blockDim.x + threadIdx.x;
  if (i >= total) return;
  int c = i % F;
  float v = (x[i] - mu[c]) * rs[c] * gamma[c] + beta[c];
  if (relu) v = fmaxf(v, 0.f);
  if (dst) dst[i] = v;
  if (bdst) bdst[i] = (__bf16)v;
  if (acc) acc[i] += v;
}

// ---------------- final combine: softmax(policy) blend -----------------------------
__global__ void k_combine(const float* __restrict__ x, const float* __restrict__ acc,
                          const float* __restrict__ policy, float* __restrict__ out,
                          int total, float invPower) {
  int i = blockIdx.x * blockDim.x + threadIdx.x;
  if (i >= total) return;
  float p0 = policy[0], p1 = policy[1];
  float m = fmaxf(p0, p1);
  float e0 = expf(p0 - m), e1 = expf(p1 - m);
  float inv = 1.0f / (e0 + e1);
  out[i] = (e0 * inv) * x[i] + (e1 * inv) * (acc[i] * invPower);
}

extern "C" void kernel_launch(void* const* d_in, const int* in_sizes, int n_in,
                              void* d_out, int out_size, void* d_ws, size_t ws_size,
                              hipStream_t stream) {
  const float* node_feat  = (const float*)d_in[0];
  const float* W1         = (const float*)d_in[1];
  const float* b1         = (const float*)d_in[2];
  const float* W2         = (const float*)d_in[3];
  const float* b2         = (const float*)d_in[4];
  const float* bn1_gamma  = (const float*)d_in[5];
  const float* bn1_beta   = (const float*)d_in[6];
  const float* strc_W     = (const float*)d_in[7];
  const float* strc_gamma = (const float*)d_in[8];
  const float* strc_beta  = (const float*)d_in[9];
  const float* policy     = (const float*)d_in[10];
  const int*   edge_index = (const int*)d_in[11];

  const int IN  = 512;
  const int HID = in_sizes[2];        // 512
  const int OUT = in_sizes[4];        // 256
  const int N   = in_sizes[0] / IN;   // 50000
  const int E   = in_sizes[11] / 2;   // 1600000
  const int* row0 = edge_index;
  const int* col0 = edge_index + E;

  // ---- workspace layout (float slots; bf16 regions use 2 elems per slot) ----
  float* wsf = (float*)d_ws;
  size_t off = 0;
  float* dis  = wsf + off; off += (size_t)N;
  float* bsum = wsf + off; off += 512;
  float* bsq  = wsf + off; off += 512;
  float* bmu  = wsf + off; off += 512;
  float* brs  = wsf + off; off += 512;
  __bf16* w1t = (__bf16*)(wsf + off); off += (size_t)IN  * HID / 2;  // [HID,IN] bf16
  __bf16* w2t = (__bf16*)(wsf + off); off += (size_t)HID * OUT / 2;  // [OUT,HID] bf16
  __bf16* abf = (__bf16*)(wsf + off); off += (size_t)N * HID / 2;    // bf16 A (X, then x1)
  float* hbuf = wsf + off; off += (size_t)N * HID;    // h1; later h2 + agg2
  float* abuf = wsf + off; off += (size_t)N * HID;    // agg1; later temp + Wc
  float* accb = wsf + off; off += (size_t)N * OUT;    // STRC accumulator

  const int TB = 256;
  const int egrid = (int)(((long long)E * 32 + TB - 1) / TB);

  auto bnrun = [&](const float* x, const float* gamma, const float* beta, float* dst,
                   __bf16* bdst, float* acc, int relu, int F) {
    k_fill<<<4, TB, 0, stream>>>(bsum, 0.f, 1024);   // zeros bsum+bsq (contiguous)
    dim3 gs((F + TB - 1) / TB, (N + 511) / 512);
    k_bn_partial<<<gs, TB, 0, stream>>>(x, bsum, bsq, N, F, 512);
    k_bn_finalize<<<(F + TB - 1) / TB, TB, 0, stream>>>(bsum, bsq, bmu, brs, N, F);
    int tot = N * F;
    k_bn_apply<<<(tot + TB - 1) / TB, TB, 0, stream>>>(x, bmu, brs, gamma, beta,
                                                       dst, bdst, acc, tot, F, relu);
  };

  // ---- degrees: deg = 1 (self loop) + count(col0); dis = deg^-1/2 ----
  k_fill<<<(N + TB - 1) / TB, TB, 0, stream>>>(dis, 1.0f, N);
  k_count_deg<<<(E + TB - 1) / TB, TB, 0, stream>>>(col0, dis, E);
  k_rsqrt_inplace<<<(N + TB - 1) / TB, TB, 0, stream>>>(dis, N);

  // ---- precision prep: X -> bf16, W1/W2 -> transposed bf16 ----
  int totX = N * IN;
  k_cvt_bf16<<<(totX + TB - 1) / TB, TB, 0, stream>>>(node_feat, abf, totX);
  k_transpose_bf16<<<(IN * HID + TB - 1) / TB, TB, 0, stream>>>(W1, w1t, IN, HID);
  k_transpose_bf16<<<(HID * OUT + TB - 1) / TB, TB, 0, stream>>>(W2, w2t, HID, OUT);

  // ---- GCN layer 1: h1 = X@W1 ; agg1 = sym-norm aggregate + b1 ; BN+ReLU -> bf16 x1
  dim3 g1((N + 127) / 128, HID / 64);
  k_gemm_bf16_wmma<<<g1, TB, 0, stream>>>(abf, w1t, hbuf, N, IN, HID);
  int tot1 = N * HID;
  k_self_bias<<<(tot1 + TB - 1) / TB, TB, 0, stream>>>(hbuf, dis, b1, abuf, tot1, HID);
  k_edge_scatter<<<egrid, TB, 0, stream>>>(hbuf, row0, col0, dis, abuf, E, HID);
  bnrun(abuf, bn1_gamma, bn1_beta, nullptr, abf, nullptr, 1, HID);  // x1 -> abf (bf16)

  // ---- GCN layer 2: h2 = x1@W2 ; agg2 = aggregate + b2 ----
  float* h2   = hbuf;
  float* agg2 = hbuf + (size_t)N * OUT;
  dim3 g2((N + 127) / 128, OUT / 64);
  k_gemm_bf16_wmma<<<g2, TB, 0, stream>>>(abf, w2t, h2, N, HID, OUT);
  int tot2 = N * OUT;
  k_self_bias<<<(tot2 + TB - 1) / TB, TB, 0, stream>>>(h2, dis, b2, agg2, tot2, OUT);
  k_edge_scatter<<<egrid, TB, 0, stream>>>(h2, row0, col0, dis, agg2, E, OUT);

  // ---- STRC branch: 2 power iterations of raw-adjacency SpMM + BN ----
  float* temp = abuf;
  float* Wc   = abuf + (size_t)N * OUT;
  k_fill<<<(tot2 + TB - 1) / TB, TB, 0, stream>>>(accb, 0.f, tot2);

  // iter 0: temp = A @ strc_W  (out[row0] += strc_W[col0])
  k_fill<<<(tot2 + TB - 1) / TB, TB, 0, stream>>>(temp, 0.f, tot2);
  k_edge_scatter<<<egrid, TB, 0, stream>>>(strc_W, col0, row0, nullptr, temp, E, OUT);
  bnrun(temp, strc_gamma, strc_beta, Wc, nullptr, accb, 0, OUT);

  // iter 1: temp = A @ Wc
  k_fill<<<(tot2 + TB - 1) / TB, TB, 0, stream>>>(temp, 0.f, tot2);
  k_edge_scatter<<<egrid, TB, 0, stream>>>(Wc, col0, row0, nullptr, temp, E, OUT);
  bnrun(temp, strc_gamma + OUT, strc_beta + OUT, Wc, nullptr, accb, 0, OUT);

  // ---- final blend: out = softmax(policy)[0]*agg2 + [1]*(accb/POWER) ----
  k_combine<<<(tot2 + TB - 1) / TB, TB, 0, stream>>>(agg2, accb, policy,
                                                     (float*)d_out, tot2, 0.5f);
}
